// RNN_20813411517075
// MI455X (gfx1250) — matrix operand order repaired and checked by reading, not compiled, not run on hardware
//
#include <hip/hip_runtime.h>

// MI455X (gfx1250) persistent-kernel RNN scan.
//   B=32, T=2000, NI=16, NH=1024, NO=16, DT/TAU = 0.1
// Recurrent GEMM per step via v_wmma_f32_16x16x32_f16 (f16 in, f32 acc).
// A-matrix (h_tr, 64KB/step) staged into LDS via global_load_async_to_lds_b128.
// Workspace layout (needs ~2.4 MB):
//   [0,256)           : grid barrier counter
//   [256, +2MiB)      : W_hh converted to f16  [1024][1024]
//   next 128 KiB      : h_tr double buffer [2][32][1024] f16
//   next 128 KiB      : z partials [2][32 wg][32 b][16 o] f32

typedef __attribute__((ext_vector_type(16))) _Float16 v16h;
typedef __attribute__((ext_vector_type(8)))  _Float16 v8h;
typedef __attribute__((ext_vector_type(4)))  _Float16 v4h;
typedef __attribute__((ext_vector_type(8)))  float    v8f;
typedef __attribute__((ext_vector_type(4)))  int      v4i;

#define NWG  32
#define TPB  128
#define Tn   2000
#define NHn  1024

// padded LDS strides (in halves) -> conflict-free ds_load_b128
#define ALDS 1032   // A tile row stride   (516 dwords -> 4-dword bank step/lane)
#define ZST  24     // zprev/xt row stride (12 dwords)
#define HST  40     // htr tile row stride (20 dwords)

#define AS1 __attribute__((address_space(1)))
#define AS3 __attribute__((address_space(3)))

#if __has_builtin(__builtin_amdgcn_global_load_async_to_lds_b128) && \
    __has_builtin(__builtin_amdgcn_s_wait_asynccnt)
#define USE_ASYNC_LDS 1
#else
#define USE_ASYNC_LDS 0
#endif

__global__ void rnn_zero_ctr(unsigned* ctr) {
  if (threadIdx.x == 0) *ctr = 0u;
}

__device__ __forceinline__ void grid_sync(unsigned* ctr, unsigned target) {
  __threadfence();            // make this WG's stores agent-visible
  __syncthreads();
  if (threadIdx.x == 0) {
    __hip_atomic_fetch_add(ctr, 1u, __ATOMIC_RELEASE, __HIP_MEMORY_SCOPE_AGENT);
    while (__hip_atomic_load(ctr, __ATOMIC_ACQUIRE, __HIP_MEMORY_SCOPE_AGENT) < target)
      __builtin_amdgcn_s_sleep(1);
  }
  __syncthreads();
  __threadfence();            // invalidate local caches on acquire side
}

__launch_bounds__(TPB, 1)
__global__ void rnn_scan(const float* __restrict__ x,      // [32][2000][16]
                         const float* __restrict__ h0,     // [32][1024]
                         const float* __restrict__ Wih,    // [1024][16]
                         const float* __restrict__ Whh,    // [1024][1024]
                         const float* __restrict__ Whz,    // [16][1024]
                         const float* __restrict__ Wzh,    // [1024][16]
                         float* __restrict__ out_z,        // [32][2000][16]
                         float* __restrict__ out_h,        // [32][2000][1024]
                         _Float16* __restrict__ whh_h,     // [1024][1024] f16
                         _Float16* __restrict__ htr_buf,   // [2][32][1024] f16
                         float* __restrict__ zpart,        // [2][NWG][512]
                         unsigned* __restrict__ ctr)
{
  const int tid  = threadIdx.x;
  const int lane = tid & 31;
  const int wave = tid >> 5;                 // 4 waves per WG
  const int half = lane >> 4;                // ISA lane-group (0: lanes 0-15)
  const int l16  = lane & 15;
  const int wg   = blockIdx.x;
  const int col0 = wg * 32;                  // this WG's 32 hidden columns
  const int m0   = (wave & 1) * 16;          // batch tile base (0 or 16)
  const int n    = col0 + (wave >> 1) * 16 + l16;  // lane's hidden column (fixed)

  __shared__ _Float16 a_lds[32 * ALDS];      // staged h_tr (A matrix), padded
  __shared__ _Float16 zprev_h[32 * ZST];     // z_prev as f16 [b][o]
  __shared__ _Float16 xt_h[32 * ZST];        // x[:,t,:] as f16 [b][i]
  __shared__ _Float16 htr_t[32 * HST];       // this WG's h_tr tile [b][c]

  // ---- time-invariant per-lane B fragments (kept in VGPRs all 2000 steps) ----
  // B_ext (32x16): rows 0..15 = W_zh^T, rows 16..31 = W_ih^T, column n.
  v16h b_ext;
  {
    const float* srcz = Wzh + (size_t)n * 16 + half * 8;
    const float* srcx = Wih + (size_t)n * 16 + half * 8;
#pragma unroll
    for (int i = 0; i < 8; i++) { b_ext[i] = (_Float16)srcz[i]; b_ext[i + 8] = (_Float16)srcx[i]; }
  }
  // b_hz (32x16) for the z-projection WMMA: B[k=c][o] = Whz[o][col0+c].
  v16h b_hz;
  {
    const float* s = Whz + (size_t)l16 * NHn + col0 + half * 16;
#pragma unroll
    for (int i = 0; i < 16; i++) b_hz[i] = (_Float16)s[i];
  }

  // ---- per-lane register h-state: 8 elems, m = m0 + i + half*8, column n ----
  float  hst[8];
  size_t hobase[8];  // out_h base offset per element
  int    bwoff[8];   // offset into h_tr global buffer
  int    ldoff[8];   // offset into LDS htr_t tile
#pragma unroll
  for (int i = 0; i < 8; i++) {
    const int m = m0 + i + half * 8;         // C/D layout: VGPR i -> M = i (+8 hi lanes)
    hst[i]    = h0[(size_t)m * NHn + n];
    hobase[i] = (size_t)m * Tn * NHn + n;
    bwoff[i]  = m * NHn + n;
    ldoff[i]  = m * HST + (n - col0);
  }

  // ---- preamble: W_hh -> f16 (vectorized), h_tr0 = tanh(h0), z0 partials ----
  for (int q = wg * TPB + tid; q < (NHn * NHn) / 4; q += NWG * TPB) {
    const float4 v = *(const float4*)(Whh + (size_t)q * 4);
    v4h o; o[0] = (_Float16)v.x; o[1] = (_Float16)v.y;
           o[2] = (_Float16)v.z; o[3] = (_Float16)v.w;
    *(v4h*)(whh_h + (size_t)q * 4) = o;
  }

#pragma unroll
  for (int i = 0; i < 8; i++) {
    const _Float16 hh = (_Float16)tanhf(hst[i]);
    htr_buf[bwoff[i]] = hh;                  // buffer slot 0
    htr_t[ldoff[i]]   = hh;
  }
  __syncthreads();

  const int wv = __builtin_amdgcn_readfirstlane(wave);  // scalar branch: EXEC all-1
  if (wv < 2) {                              // z0 partial = htr_tile @ Whz_slice^T
    const int mz0 = wv * 16;
    const v8h a0 = *(const v8h*)&htr_t[(mz0 + l16) * HST + half * 16];
    const v8h a1 = *(const v8h*)&htr_t[(mz0 + l16) * HST + half * 16 + 8];
    v16h a;
#pragma unroll
    for (int i = 0; i < 8; i++) { a[i] = a0[i]; a[i + 8] = a1[i]; }
    v8f c = {};
    v8f z = __builtin_amdgcn_wmma_f32_16x16x32_f16(false, a, false, b_hz,
                                                   (short)0, c, false, false);
    float* zp0 = zpart + (size_t)wg * 512;   // parity-0 slot
#pragma unroll
    for (int i = 0; i < 8; i++) zp0[(mz0 + i + half * 8) * 16 + l16] = z[i];
  }

  unsigned epoch = 1;
  grid_sync(ctr, epoch * NWG); epoch++;

  // =========================== sequential scan ===========================
  for (int t = 0; t < Tn; t++) {
    const _Float16* bufR = htr_buf + (size_t)(t & 1) * 32 * NHn;
    _Float16*       bufW = htr_buf + (size_t)((t + 1) & 1) * 32 * NHn;
    const float*    zpR  = zpart + (size_t)(t & 1) * NWG * 512;
    float*          zpW  = zpart + (size_t)((t + 1) & 1) * NWG * 512 + (size_t)wg * 512;

    // phase 0: stage full h_tr (A matrix, 64KB) into LDS.
    // 4096 chunks of 16B; 128 threads x 32 chunks. Async DMA engine,
    // overlapped with the z-reduction / x staging below.
#pragma unroll 4
    for (int r = 0; r < 32; r++) {
      const int c  = r * TPB + tid;
      const int m  = c >> 7;                 // 128 chunks per 1024-half row
      const int k8 = (c & 127) * 8;
#if USE_ASYNC_LDS
      __builtin_amdgcn_global_load_async_to_lds_b128(
          (AS1 v4i*)(uintptr_t)(bufR + (size_t)m * NHn + k8),
          (AS3 v4i*)(unsigned)(uintptr_t)(&a_lds[m * ALDS + k8]),
          0, 0);
#else
      const v8h tmp = *(const v8h*)(bufR + (size_t)m * NHn + k8);
      *(v8h*)(&a_lds[m * ALDS + k8]) = tmp;
#endif
    }

    // phase 1: deterministic reduce of z partials -> LDS f16; WG0 emits z_{t-1}.
    // phase 2: stage x[:,t,:] to LDS f16.
    {
      float4 s = make_float4(0.f, 0.f, 0.f, 0.f);
#pragma unroll 4
      for (int w = 0; w < NWG; w++) {
        const float4 p = *(const float4*)(zpR + (size_t)w * 512 + tid * 4);
        s.x += p.x; s.y += p.y; s.z += p.z; s.w += p.w;
      }
      const int b = tid >> 2, o0 = (tid * 4) & 15;
      _Float16* zh = &zprev_h[b * ZST + o0];
      zh[0] = (_Float16)s.x; zh[1] = (_Float16)s.y;
      zh[2] = (_Float16)s.z; zh[3] = (_Float16)s.w;
      if (wg == 0 && t > 0)
        *(float4*)(out_z + (size_t)b * Tn * 16 + (size_t)(t - 1) * 16 + o0) = s;

      const float4 xv = *(const float4*)(x + (size_t)b * Tn * 16 + (size_t)t * 16 + o0);
      _Float16* xh = &xt_h[b * ZST + o0];
      xh[0] = (_Float16)xv.x; xh[1] = (_Float16)xv.y;
      xh[2] = (_Float16)xv.z; xh[3] = (_Float16)xv.w;
    }
#if USE_ASYNC_LDS
    __builtin_amdgcn_s_wait_asynccnt(0);     // A tile landed in LDS
#endif
    __syncthreads();

    // phase 3: acc = z_prev@Wzh^T + x_t@Wih^T via ONE wmma (A = [z_prev | x_t])
    v8f acc;
    {
      const int ma = m0 + l16;
      const v8h z8 = *(const v8h*)&zprev_h[ma * ZST + half * 8];
      const v8h x8 = *(const v8h*)&xt_h[ma * ZST + half * 8];
      v16h a;
#pragma unroll
      for (int i = 0; i < 8; i++) { a[i] = z8[i]; a[i + 8] = x8[i]; }
      v8f c = {};
      acc = __builtin_amdgcn_wmma_f32_16x16x32_f16(false, a, false, b_ext,
                                                   (short)0, c, false, false);
    }

    // phase 4: recurrent GEMM, K=1024 -> 32 x wmma_f32_16x16x32_f16.
    // A fragments from LDS (conflict-free), B streamed from L2-resident f16 W_hh.
    {
      const _Float16* aBase = a_lds + (m0 + l16) * ALDS + half * 16;
      const _Float16* bBase = whh_h + (size_t)n * NHn + half * 16;
#pragma unroll 4
      for (int kc = 0; kc < 32; kc++) {
        const int k0 = kc * 32;
        const v8h a0 = *(const v8h*)(aBase + k0);        // A: K = k0+h*16 .. +8
        const v8h a1 = *(const v8h*)(aBase + k0 + 8);    // A: K = k0+h*16+8 .. +8
        const v16h bf = *(const v16h*)(bBase + k0);      // B col n, 16 contiguous K
        __builtin_prefetch((const void*)(bBase + k0 + 512), 0, 0);  // global_prefetch
        v16h a;
#pragma unroll
        for (int i = 0; i < 8; i++) { a[i] = a0[i]; a[i + 8] = a1[i]; }
        acc = __builtin_amdgcn_wmma_f32_16x16x32_f16(false, a, false, bf,
                                                     (short)0, acc, false, false);
      }
    }

    // phase 5: h_t = 0.9*h + 0.1*(rec + xp + zW); emit h; h_tr -> buffers
    const size_t tOff = (size_t)t * NHn;
#pragma unroll
    for (int i = 0; i < 8; i++) {
      const float hn = 0.9f * hst[i] + 0.1f * acc[i];
      hst[i] = hn;
      __builtin_nontemporal_store(hn, out_h + hobase[i] + tOff);
      const _Float16 hh = (_Float16)tanhf(hn);
      bufW[bwoff[i]]  = hh;
      htr_t[ldoff[i]] = hh;
    }
    __syncthreads();

    // phase 6: z partial for step t via wmma (waves 0,1; scalar-uniform branch)
    if (wv < 2) {
      const int mz0 = wv * 16;
      const v8h a0 = *(const v8h*)&htr_t[(mz0 + l16) * HST + half * 16];
      const v8h a1 = *(const v8h*)&htr_t[(mz0 + l16) * HST + half * 16 + 8];
      v16h a;
#pragma unroll
      for (int i = 0; i < 8; i++) { a[i] = a0[i]; a[i + 8] = a1[i]; }
      v8f c = {};
      v8f z = __builtin_amdgcn_wmma_f32_16x16x32_f16(false, a, false, b_hz,
                                                     (short)0, c, false, false);
#pragma unroll
      for (int i = 0; i < 8; i++) zpW[(mz0 + i + half * 8) * 16 + l16] = z[i];
    }

    grid_sync(ctr, epoch * NWG); epoch++;
  }

  // epilogue: z_{T-1} lives in parity slot (Tn & 1) == 0; WG0 reduces + writes.
  if (wg == 0) {
    const float* zpR = zpart + (size_t)(Tn & 1) * NWG * 512;
    float4 s = make_float4(0.f, 0.f, 0.f, 0.f);
#pragma unroll 4
    for (int w = 0; w < NWG; w++) {
      const float4 p = *(const float4*)(zpR + (size_t)w * 512 + tid * 4);
      s.x += p.x; s.y += p.y; s.z += p.z; s.w += p.w;
    }
    const int b = tid >> 2, o0 = (tid * 4) & 15;
    *(float4*)(out_z + (size_t)b * Tn * 16 + (size_t)(Tn - 1) * 16 + o0) = s;
  }
}

extern "C" void kernel_launch(void* const* d_in, const int* in_sizes, int n_in,
                              void* d_out, int out_size, void* d_ws, size_t ws_size,
                              hipStream_t stream) {
  const float* x   = (const float*)d_in[0];
  const float* h0  = (const float*)d_in[1];
  const float* Wih = (const float*)d_in[2];
  const float* Whh = (const float*)d_in[3];
  const float* Whz = (const float*)d_in[4];
  const float* Wzh = (const float*)d_in[5];

  float* out_z = (float*)d_out;                       // [32][2000][16]
  float* out_h = out_z + (size_t)32 * Tn * 16;        // [32][2000][1024]

  uintptr_t ws = (uintptr_t)d_ws;
  unsigned*  ctr     = (unsigned*)ws;
  _Float16*  whh_h   = (_Float16*)(ws + 256);
  _Float16*  htr_buf = (_Float16*)(ws + 256 + 2097152);
  float*     zpart   = (float*)(ws + 256 + 2097152 + 131072);
  (void)ws_size; (void)in_sizes; (void)n_in; (void)out_size;

  hipLaunchKernelGGL(rnn_zero_ctr, dim3(1), dim3(64), 0, stream, ctr);
  hipLaunchKernelGGL(rnn_scan, dim3(NWG), dim3(TPB), 0, stream,
                     x, h0, Wih, Whh, Whz, Wzh, out_z, out_h,
                     whh_h, htr_buf, zpart, ctr);
}
